// DQNGNN_31851477467869
// MI455X (gfx1250) — compile-verified, matrix operand order checked by reading, not run on hardware
//
#include <hip/hip_runtime.h>
#include <hip/hip_bf16.h>

// ---------------------------------------------------------------------------
// DQN-GNN forward for MI455X (gfx1250, wave32).
// Dominant GEMM (conv1 edge-MLP layer2 fused with per-edge contraction):
//   msg[e,o] = sum_{k,i} hid1[e,k] * hsrc[e,i] * e2_w1[k, i*64+o]
// computed as U[E,8192] @ B[8192,64] with U built on the fly (rank-1 per edge),
// using v_wmma_f32_16x16x32_bf16 (f32 accumulate). B pre-swizzled to fragment
// order; 32-edge M-tile gives 2x B reuse; D fragments accumulated via
// ds_add_f32 into a small LDS tile. Pair MLP also WMMA.
// ---------------------------------------------------------------------------

#define N_NODES 10000
#define N_EDGES 50000
#define N_PIPES 20000
#define NODE_IN 4
#define H 64
#define LN_EPS 1e-5f

typedef __attribute__((ext_vector_type(16))) __bf16 v16bf;
typedef __attribute__((ext_vector_type(8)))  float  v8f;

// --------------------------------- utils -----------------------------------

__global__ void k_zero(float* p, int n) {
    int i = blockIdx.x * blockDim.x + threadIdx.x;
    if (i < n) p[i] = 0.0f;
}

// ----------------------- edge MLP layer 1 (relu) ----------------------------
template<int OC>
__global__ void k_edge_mlp1(const float* __restrict__ ea,
                            const float* __restrict__ w,
                            const float* __restrict__ b,
                            float* __restrict__ hid) {
    int idx = blockIdx.x * blockDim.x + threadIdx.x;
    if (idx >= N_EDGES * OC) return;
    int e = idx / OC, o = idx % OC;
    float a0 = ea[e * 4 + 0], a1 = ea[e * 4 + 1], a2 = ea[e * 4 + 2], a3 = ea[e * 4 + 3];
    float acc = b[o];
    acc += a0 * w[0 * OC + o];
    acc += a1 * w[1 * OC + o];
    acc += a2 * w[2 * OC + o];
    acc += a3 * w[3 * OC + o];
    hid[idx] = fmaxf(acc, 0.0f);
}

// -------------------- conv0 message + scatter (VALU, cheap) -----------------
__global__ void k_msg0(const float* __restrict__ x,
                       const int* __restrict__ src,
                       const int* __restrict__ dst,
                       const float* __restrict__ hid0,
                       const float* __restrict__ e2w,   // [64, 256]
                       const float* __restrict__ e2b,   // [256]
                       float* __restrict__ agg,         // [N,64]
                       float* __restrict__ cnt) {       // [N]
    int idx = blockIdx.x * blockDim.x + threadIdx.x;
    if (idx >= N_EDGES * H) return;
    int e = idx >> 6, o = idx & 63;
    int s = src[e];
    float xs[4];
#pragma unroll
    for (int i = 0; i < 4; ++i) xs[i] = x[s * 4 + i];
    float wi[4] = {0.f, 0.f, 0.f, 0.f};
    for (int k = 0; k < 64; ++k) {
        float hk = hid0[e * 64 + k];
#pragma unroll
        for (int i = 0; i < 4; ++i) wi[i] += hk * e2w[k * 256 + i * 64 + o];
    }
    float m = 0.f;
#pragma unroll
    for (int i = 0; i < 4; ++i) m += xs[i] * (wi[i] + e2b[i * 64 + o]);
    atomicAdd(&agg[dst[e] * 64 + o], m);
    if (o == 0) atomicAdd(&cnt[dst[e]], 1.0f);
}

// ------------------- node update: root GEMV + mean + relu + LN --------------
template<int IN>
__global__ __launch_bounds__(64)
void k_node(const float* __restrict__ hprev,
            const float* __restrict__ agg,
            const float* __restrict__ cnt,
            const float* __restrict__ root,   // [IN, 64]
            const float* __restrict__ bias,
            const float* __restrict__ ln_g,
            const float* __restrict__ ln_b,
            float* __restrict__ hout) {
    int n = blockIdx.x;
    int o = threadIdx.x;
    __shared__ float s_in[IN];
    __shared__ float red[64];
    for (int k = o; k < IN; k += 64) s_in[k] = hprev[n * IN + k];
    __syncthreads();
    float acc = bias[o];
    for (int k = 0; k < IN; ++k) acc += s_in[k] * root[k * 64 + o];
    float c = fmaxf(cnt[n], 1.0f);
    acc += agg[n * 64 + o] / c;
    acc = fmaxf(acc, 0.0f);
    red[o] = acc; __syncthreads();
    for (int s = 32; s > 0; s >>= 1) { if (o < s) red[o] += red[o + s]; __syncthreads(); }
    float mu = red[0] * (1.0f / 64.0f); __syncthreads();
    float d = acc - mu;
    red[o] = d * d; __syncthreads();
    for (int s = 32; s > 0; s >>= 1) { if (o < s) red[o] += red[o + s]; __syncthreads(); }
    float var = red[0] * (1.0f / 64.0f);
    hout[n * 64 + o] = d * rsqrtf(var + LN_EPS) * ln_g[o] + ln_b[o];
}

// ----------------- B-matrix pre-swizzle into WMMA fragment order ------------
// lane holds column n = ntile*16+(lane&15); element j -> K' =
// kstep*32 + (j>>3)*16 + (lane>=16 ? 8 : 0) + (j&7).
// Source B1[kk, o] = e2_w1[kk>>6, (kk&63)*64 + o]   (kk = k*64 + i)
__global__ void k_packB1(const float* __restrict__ e2w, __bf16* __restrict__ packed) {
    int idx = blockIdx.x * blockDim.x + threadIdx.x;      // 256 ksteps * 4 tiles * 32 * 16
    if (idx >= 256 * 4 * 32 * 16) return;
    int j = idx & 15, lane = (idx >> 4) & 31, t = (idx >> 9) & 3, kstep = idx >> 11;
    int kk = kstep * 32 + ((j >> 3) << 4) + ((lane >= 16) ? 8 : 0) + (j & 7);
    int k = kk >> 6, i = kk & 63;
    int n = t * 16 + (lane & 15);
    packed[idx] = (__bf16)e2w[k * 4096 + i * 64 + n];
}

__global__ void k_packBp(const float* __restrict__ pw, __bf16* __restrict__ packed) {
    int idx = blockIdx.x * blockDim.x + threadIdx.x;      // 4 ksteps * 4 tiles * 32 * 16
    if (idx >= 4 * 4 * 32 * 16) return;
    int j = idx & 15, lane = (idx >> 4) & 31, t = (idx >> 9) & 3, kstep = idx >> 11;
    int kk = kstep * 32 + ((j >> 3) << 4) + ((lane >= 16) ? 8 : 0) + (j & 7);
    int n = t * 16 + (lane & 15);
    packed[idx] = (__bf16)pw[kk * 64 + n];
}

// --------------- conv1 fused message kernel (the 52-GFLOP GEMM) -------------
// Workgroup = 256 threads (8 waves), 32 edges (2 M-subtiles => 2x B reuse).
// Each wave owns 32 K-steps; per K-step: load 4 B tiles, build 2 A fragments,
// issue 8 v_wmma_f32_16x16x32_bf16. D fragments accumulated with ds_add_f32
// into s_msg; one pass adds the e2_b bias term and scatters via global atomics.
__global__ __launch_bounds__(256)
void k_msg1(const float* __restrict__ hid1,     // [E,128]
            const float* __restrict__ h1,       // [N,64]
            const int* __restrict__ src,
            const int* __restrict__ dst,
            const __bf16* __restrict__ packB,   // fragment-packed [8192,64]
            const float* __restrict__ e2b,      // [4096]
            float* __restrict__ agg) {          // [N,64]
    __shared__ float s_hid[32 * 128];           // 16 KB
    __shared__ float s_hsrc[32 * 64];           // 8 KB
    __shared__ float s_msg[32 * 64];            // 8 KB
    const int e0 = blockIdx.x * 32;
    const int tid = threadIdx.x;

    for (int idx = tid; idx < 32 * 128; idx += 256) {
        int m = idx >> 7, k = idx & 127;
        s_hid[idx] = (e0 + m < N_EDGES) ? hid1[(size_t)(e0 + m) * 128 + k] : 0.0f;
    }
    for (int idx = tid; idx < 32 * 64; idx += 256) {
        int m = idx >> 6, i = idx & 63;
        s_hsrc[idx] = (e0 + m < N_EDGES) ? h1[(size_t)src[e0 + m] * 64 + i] : 0.0f;
        s_msg[idx] = 0.0f;
    }
    __syncthreads();

    const int wave = tid >> 5, lane = tid & 31;
    const int mrow = lane & 15;
    const int hi8 = (lane >= 16) ? 8 : 0;
    v8f acc[2][4] = {{v8f{}, v8f{}, v8f{}, v8f{}}, {v8f{}, v8f{}, v8f{}, v8f{}}};

    for (int s = 0; s < 32; ++s) {
        const int kstep = wave * 32 + s;        // 0..255 over K' = 8192
        // load all 4 B tiles first (latency overlapped by A build below)
        const __bf16* bptr = packB + (size_t)kstep * 2048 + lane * 16;
        v16bf b0 = *(const v16bf*)(bptr + 0 * 512);
        v16bf b1 = *(const v16bf*)(bptr + 1 * 512);
        v16bf b2 = *(const v16bf*)(bptr + 2 * 512);
        v16bf b3 = *(const v16bf*)(bptr + 3 * 512);
        // build rank-1 A fragments for both M-subtiles
        v16bf a0, a1;
#pragma unroll
        for (int j = 0; j < 16; ++j) {
            int kk = kstep * 32 + ((j >> 3) << 4) + hi8 + (j & 7);
            int k = kk >> 6, i = kk & 63;
            a0[j] = (__bf16)(s_hid[mrow * 128 + k] * s_hsrc[mrow * 64 + i]);
            a1[j] = (__bf16)(s_hid[(16 + mrow) * 128 + k] * s_hsrc[(16 + mrow) * 64 + i]);
        }
        acc[0][0] = __builtin_amdgcn_wmma_f32_16x16x32_bf16(false, a0, false, b0, (short)0, acc[0][0], false, false);
        acc[0][1] = __builtin_amdgcn_wmma_f32_16x16x32_bf16(false, a0, false, b1, (short)0, acc[0][1], false, false);
        acc[0][2] = __builtin_amdgcn_wmma_f32_16x16x32_bf16(false, a0, false, b2, (short)0, acc[0][2], false, false);
        acc[0][3] = __builtin_amdgcn_wmma_f32_16x16x32_bf16(false, a0, false, b3, (short)0, acc[0][3], false, false);
        acc[1][0] = __builtin_amdgcn_wmma_f32_16x16x32_bf16(false, a1, false, b0, (short)0, acc[1][0], false, false);
        acc[1][1] = __builtin_amdgcn_wmma_f32_16x16x32_bf16(false, a1, false, b1, (short)0, acc[1][1], false, false);
        acc[1][2] = __builtin_amdgcn_wmma_f32_16x16x32_bf16(false, a1, false, b2, (short)0, acc[1][2], false, false);
        acc[1][3] = __builtin_amdgcn_wmma_f32_16x16x32_bf16(false, a1, false, b3, (short)0, acc[1][3], false, false);
    }

    // accumulate D fragments into LDS tile via ds_add_f32 (no intra-wave conflicts)
#pragma unroll
    for (int mt = 0; mt < 2; ++mt)
#pragma unroll
        for (int t = 0; t < 4; ++t)
#pragma unroll
            for (int r = 0; r < 8; ++r) {
                int m = mt * 16 + r + hi8;
                int n = t * 16 + (lane & 15);
                atomicAdd(&s_msg[m * 64 + n], acc[mt][t][r]);
            }
    __syncthreads();

    // bias term + scatter-add to agg
    for (int idx = tid; idx < 32 * 64; idx += 256) {
        int m = idx >> 6, o = idx & 63;
        int e = e0 + m;
        if (e < N_EDGES) {
            float bterm = 0.0f;
            for (int i = 0; i < 64; ++i) bterm += s_hsrc[m * 64 + i] * e2b[i * 64 + o];
            atomicAdd(&agg[(size_t)dst[e] * 64 + o], s_msg[idx] + bterm);
        }
    }
}

// ----------------------------- global mean pool -----------------------------
__global__ __launch_bounds__(64)
void k_pool(const float* __restrict__ h2, float* __restrict__ gvec) {
    int o = threadIdx.x;
    float s = 0.0f;
    for (int n = 0; n < N_NODES; ++n) s += h2[n * 64 + o];
    gvec[o] = s / (float)N_NODES;
}

// ------------------------ global head: gh MLP + q_global --------------------
__global__ __launch_bounds__(64)
void k_ghead(const float* __restrict__ gvec,
             const float* __restrict__ gh_w, const float* __restrict__ gh_b,
             const float* __restrict__ ln_g, const float* __restrict__ ln_b,
             const float* __restrict__ qg_w, const float* __restrict__ qg_b,
             float* __restrict__ qglob, float* __restrict__ out) {
    int o = threadIdx.x;
    __shared__ float red[64];
    __shared__ float s_gh[64];
    float acc = gh_b[o];
    for (int k = 0; k < 64; ++k) acc += gvec[k] * gh_w[k * 64 + o];
    acc = fmaxf(acc, 0.0f);
    red[o] = acc; __syncthreads();
    for (int s = 32; s > 0; s >>= 1) { if (o < s) red[o] += red[o + s]; __syncthreads(); }
    float mu = red[0] * (1.0f / 64.0f); __syncthreads();
    float d = acc - mu;
    red[o] = d * d; __syncthreads();
    for (int s = 32; s > 0; s >>= 1) { if (o < s) red[o] += red[o + s]; __syncthreads(); }
    float var = red[0] * (1.0f / 64.0f);
    s_gh[o] = d * rsqrtf(var + LN_EPS) * ln_g[o] + ln_b[o];
    __syncthreads();
    if (o < 2) {
        float q = qg_b[o];
        for (int k = 0; k < 64; ++k) q += s_gh[k] * qg_w[k * 2 + o];
        qglob[o] = q;
        out[2 * N_PIPES + o] = q;
    }
}

// ------------------- pair head: WMMA GEMM + LN + qn + mask ------------------
__global__ __launch_bounds__(128)
void k_pair(const float* __restrict__ h2,
            const int* __restrict__ src, const int* __restrict__ dst,
            const int* __restrict__ pidx, const float* __restrict__ open_mask,
            const __bf16* __restrict__ packBp,
            const float* __restrict__ pair_b,
            const float* __restrict__ ln_g, const float* __restrict__ ln_b,
            const float* __restrict__ qn_w, const float* __restrict__ qn_b,
            const float* __restrict__ qglob,
            float* __restrict__ out) {
    __shared__ float s_pair[16 * 128];
    __shared__ float s_feat[16 * 64];
    const int p0 = blockIdx.x * 16;
    const int tid = threadIdx.x;

    for (int idx = tid; idx < 16 * 128; idx += 128) {
        int m = idx >> 7, c = idx & 127;
        float v = 0.0f;
        int p = p0 + m;
        if (p < N_PIPES) {
            int e = pidx[p];
            int node = (c < 64) ? src[e] : dst[e];
            v = h2[node * 64 + (c & 63)];
        }
        s_pair[idx] = v;
    }
    __syncthreads();

    const int wave = tid >> 5, lane = tid & 31;   // wave == ntile
    const int mrow = lane & 15;
    const int khalf = (lane >= 16) ? 8 : 0;
    v8f acc = v8f{};
    for (int kstep = 0; kstep < 4; ++kstep) {
        v16bf a;
#pragma unroll
        for (int j = 0; j < 16; ++j) {
            int kk = kstep * 32 + ((j >> 3) << 4) + khalf + (j & 7);
            a[j] = (__bf16)s_pair[mrow * 128 + kk];
        }
        const v16bf b = *(const v16bf*)(packBp + (((kstep * 4 + wave) * 32) + lane) * 16);
        acc = __builtin_amdgcn_wmma_f32_16x16x32_bf16(
            false, a, false, b, (short)0, acc, false, false);
    }
#pragma unroll
    for (int r = 0; r < 8; ++r) {
        int m = r + ((lane >= 16) ? 8 : 0);
        int n = wave * 16 + (lane & 15);
        s_feat[m * 64 + n] = fmaxf(acc[r] + pair_b[n], 0.0f);
    }
    __syncthreads();

    if (tid < 16) {
        int p = p0 + tid;
        if (p < N_PIPES) {
            float mu = 0.0f;
            for (int o = 0; o < 64; ++o) mu += s_feat[tid * 64 + o];
            mu *= (1.0f / 64.0f);
            float var = 0.0f;
            for (int o = 0; o < 64; ++o) { float d = s_feat[tid * 64 + o] - mu; var += d * d; }
            var *= (1.0f / 64.0f);
            float rstd = rsqrtf(var + LN_EPS);
            float q0 = qn_b[0] + qglob[0];
            float q1 = qn_b[1] + qglob[1];
            for (int o = 0; o < 64; ++o) {
                float f = (s_feat[tid * 64 + o] - mu) * rstd * ln_g[o] + ln_b[o];
                q0 += f * qn_w[o * 2 + 0];
                q1 += f * qn_w[o * 2 + 1];
            }
            float open = open_mask[p];
            out[p * 2 + 0] = (open < 0.5f) ? -1e9f : q0;
            out[p * 2 + 1] = ((1.0f - open) < 0.5f) ? -1e9f : q1;
        }
    }
}

// --------------------------------- launch -----------------------------------

extern "C" void kernel_launch(void* const* d_in, const int* in_sizes, int n_in,
                              void* d_out, int out_size, void* d_ws, size_t ws_size,
                              hipStream_t stream) {
    (void)in_sizes; (void)n_in; (void)out_size; (void)ws_size;
    // Input order: jax tree_leaves of setup_inputs() dict (sorted keys, nested).
    const float* ea       = (const float*)d_in[0];
    const int*   eidx     = (const int*)d_in[1];
    const float* c0_bias  = (const float*)d_in[2];
    const float* c0_e1b   = (const float*)d_in[3];
    const float* c0_e1w   = (const float*)d_in[4];
    const float* c0_e2b   = (const float*)d_in[5];
    const float* c0_e2w   = (const float*)d_in[6];
    const float* c0_lnb   = (const float*)d_in[7];
    const float* c0_lng   = (const float*)d_in[8];
    const float* c0_root  = (const float*)d_in[9];
    const float* c1_bias  = (const float*)d_in[10];
    const float* c1_e1b   = (const float*)d_in[11];
    const float* c1_e1w   = (const float*)d_in[12];
    const float* c1_e2b   = (const float*)d_in[13];
    const float* c1_e2w   = (const float*)d_in[14];
    const float* c1_lnb   = (const float*)d_in[15];
    const float* c1_lng   = (const float*)d_in[16];
    const float* c1_root  = (const float*)d_in[17];
    const float* gh_b     = (const float*)d_in[18];
    const float* gh_lnb   = (const float*)d_in[19];
    const float* gh_lng   = (const float*)d_in[20];
    const float* gh_w     = (const float*)d_in[21];
    const float* pair_b   = (const float*)d_in[22];
    const float* pair_lnb = (const float*)d_in[23];
    const float* pair_lng = (const float*)d_in[24];
    const float* pair_w   = (const float*)d_in[25];
    const float* qg_b     = (const float*)d_in[26];
    const float* qg_w     = (const float*)d_in[27];
    const float* qn_b     = (const float*)d_in[28];
    const float* qn_w     = (const float*)d_in[29];
    const int*   pidx     = (const int*)d_in[30];
    const float* pmask    = (const float*)d_in[31];
    const float* x        = (const float*)d_in[32];

    const int* src = eidx;
    const int* dst = eidx + N_EDGES;
    float* out = (float*)d_out;

    char* ws = (char*)d_ws;
    size_t off = 0;
    auto carve = [&](size_t bytes) -> void* {
        void* p = ws + off;
        off += (bytes + 255) & ~(size_t)255;
        return p;
    };
    float*  agg0   = (float*)carve((size_t)N_NODES * 64 * 4);
    float*  agg1   = (float*)carve((size_t)N_NODES * 64 * 4);
    float*  cnt    = (float*)carve((size_t)N_NODES * 4);
    float*  h1     = (float*)carve((size_t)N_NODES * 64 * 4);
    float*  h2     = (float*)carve((size_t)N_NODES * 64 * 4);
    float*  hid0   = (float*)carve((size_t)N_EDGES * 64 * 4);
    float*  hid1   = (float*)carve((size_t)N_EDGES * 128 * 4);
    float*  gvec   = (float*)carve(64 * 4);
    float*  qglob  = (float*)carve(64 * 4);
    __bf16* packB1 = (__bf16*)carve((size_t)8192 * 64 * 2);
    __bf16* packBp = (__bf16*)carve((size_t)128 * 64 * 2);

    {
        int nz = N_NODES * 64;
        k_zero<<<(nz + 255) / 256, 256, 0, stream>>>(agg0, nz);
        k_zero<<<(nz + 255) / 256, 256, 0, stream>>>(agg1, nz);
        k_zero<<<(N_NODES + 255) / 256, 256, 0, stream>>>(cnt, N_NODES);
    }

    k_packB1<<<(256 * 4 * 32 * 16) / 256, 256, 0, stream>>>(c1_e2w, packB1);
    k_packBp<<<(4 * 4 * 32 * 16) / 256, 256, 0, stream>>>(pair_w, packBp);

    // ---- conv0 ----
    k_edge_mlp1<64><<<(N_EDGES * 64 + 255) / 256, 256, 0, stream>>>(ea, c0_e1w, c0_e1b, hid0);
    k_msg0<<<(N_EDGES * 64 + 255) / 256, 256, 0, stream>>>(x, src, dst, hid0, c0_e2w, c0_e2b,
                                                           agg0, cnt);
    k_node<NODE_IN><<<N_NODES, 64, 0, stream>>>(x, agg0, cnt, c0_root, c0_bias, c0_lng, c0_lnb, h1);

    // ---- conv1 (WMMA-dominant) ----
    k_edge_mlp1<128><<<(N_EDGES * 128 + 255) / 256, 256, 0, stream>>>(ea, c1_e1w, c1_e1b, hid1);
    k_msg1<<<(N_EDGES + 31) / 32, 256, 0, stream>>>(hid1, h1, src, dst, packB1, c1_e2b, agg1);
    k_node<H><<<N_NODES, 64, 0, stream>>>(h1, agg1, cnt, c1_root, c1_bias, c1_lng, c1_lnb, h2);

    // ---- heads ----
    k_pool<<<1, 64, 0, stream>>>(h2, gvec);
    k_ghead<<<1, 64, 0, stream>>>(gvec, gh_w, gh_b, gh_lng, gh_lnb, qg_w, qg_b, qglob, out);
    k_pair<<<N_PIPES / 16, 128, 0, stream>>>(h2, src, dst, pidx, pmask, packBp, pair_b,
                                             pair_lng, pair_lnb, qn_w, qn_b, qglob, out);
}
